// GNNMapper_17583596110615
// MI455X (gfx1250) — matrix operand order repaired and checked by reading, not compiled, not run on hardware
//
#include <hip/hip_runtime.h>

// ---------------------------------------------------------------------------
// GNNMapper forward for MI455X (gfx1250, wave32, WMMA bf16 16x16x32).
//
//  * All dense GEMMs on v_wmma_f32_16x16x32_bf16 (bf16 in, f32 accum).
//  * Weights pre-transposed to bf16 [N][K] once per launch -> B staging is a
//    single contiguous 16B copy per lane (no LDS scalar transpose stores).
//  * bf16-A GEMMs use GLOBAL_LOAD_ASYNC_TO_LDS_B128 with a double-buffered
//    LDS pipeline (ASYNCcnt tracked, s_wait_asynccnt + barrier).
//  * Edge-MLP layer-2 algebraically fused into lin_edge:
//      ee = (h@w2+b2)@we+be == h@(w2@we) + (b2@we+be); W_ee^T computed
//    on-device as we^T @ w2^T so it lands pre-transposed.
//  * Segment softmax via L2-resident gathers (q/k/v fit in 192MB L2) and
//    f32 atomics for max/sum/scatter.
// ---------------------------------------------------------------------------

typedef __bf16 bf16;
typedef __bf16 v16bf __attribute__((ext_vector_type(16)));
typedef float  v8f   __attribute__((ext_vector_type(8)));

__device__ __forceinline__ bf16 f2bf(float f) {
  union { float f; unsigned u; } v; v.f = f;
  unsigned r = (v.u + 0x7FFFu + ((v.u >> 16) & 1u)) >> 16;  // round-nearest-even
  union { unsigned short s; bf16 b; } o; o.s = (unsigned short)r;
  return o.b;
}
__device__ __forceinline__ float bf2f(bf16 b) {
  union { unsigned short s; bf16 b; } i; i.b = b;
  union { unsigned u; float f; } o; o.u = ((unsigned)i.s) << 16;
  return o.f;
}
// Ordered-uint encoding so unsigned atomicMax == float max.
__device__ __forceinline__ unsigned f2key(float f) {
  unsigned u = __float_as_uint(f);
  return (u & 0x80000000u) ? ~u : (u | 0x80000000u);
}
__device__ __forceinline__ float key2f(unsigned k) {
  unsigned u = (k & 0x80000000u) ? (k & 0x7FFFFFFFu) : ~k;
  return __uint_as_float(u);
}
__device__ __forceinline__ float silu(float x) { return x / (1.0f + __expf(-x)); }

// ---- CDNA5 async global->LDS copy (ASYNCcnt-tracked, §10.4 / §15.18) ------
// LDS aperture maps addr[31:0] -> LDS offset, so the low 32 bits of a flat
// pointer to a __shared__ object are the LDS address VGPR operand.
__device__ __forceinline__ void async_copy_b128(const void* g, void* lds) {
  unsigned l = (unsigned)(uintptr_t)lds;
  asm volatile("global_load_async_to_lds_b128 %0, %1, off"
               :: "v"(l), "v"(g) : "memory");
}
__device__ __forceinline__ void s_wait_asynccnt0() {
  asm volatile("s_wait_asynccnt 0x0" ::: "memory");
}

// ---------------------------------------------------------------------------
// f32 -> bf16 conversions (weights)
// ---------------------------------------------------------------------------
__global__ void cvt_f32_bf16_kernel(const float* __restrict__ in,
                                    bf16* __restrict__ out, int n) {
  int i = blockIdx.x * 256 + threadIdx.x;
  if (i < n) out[i] = f2bf(in[i]);
}
// in[K][N] f32 -> out[N][K] bf16 (pre-transposed weight for GEMM B operand)
__global__ void cvt_t_f32_bf16_kernel(const float* __restrict__ in,
                                      bf16* __restrict__ out, int K, int N) {
  int i = blockIdx.x * 256 + threadIdx.x;
  if (i < K * N) {
    int k = i / N, n = i - k * N;
    out[(size_t)n * K + k] = f2bf(in[i]);
  }
}

// b_ee[n] = be[n] + sum_k b2[k] * we[k][n]   (single block, N=256)
__global__ void bias_fuse_kernel(const float* __restrict__ b2,
                                 const float* __restrict__ we,
                                 const float* __restrict__ be,
                                 float* __restrict__ bee) {
  int n = threadIdx.x;
  float acc = be[n];
  for (int k = 0; k < 256; ++k) acc += b2[k] * we[k * 256 + n];
  bee[n] = acc;
}

// ---------------------------------------------------------------------------
// Generic WMMA GEMM:  Out[M,N] = epilogue( A[M,K] @ B[K,N] ), BT = B^T [N][K].
//   A_F32   : A is f32 (converted to bf16 while staging; guarded, sync);
//             else A is bf16 in d_ws (async double-buffered, unguarded —
//             tail overreads stay inside d_ws).
//   ADD2    : A := A + A2 (both f32) before conversion.
//   SILU    : SiLU after bias.   RES_ADD : add res[M,N] f32 after activation.
//   OUT_BF16: store bf16, else f32.
// Block 256 = 8 waves (4x2); wave tile 32x32 = 2x2 WMMA accums.
// BM=128, BN=64, BK=32; LDS row stride 40 bf16 = 80B (16B-aligned,
// 20-bank stride -> conflict-free 16-lane fragment reads).
// ---------------------------------------------------------------------------
template <bool A_F32, bool ADD2, bool SILU_EP, bool RES_ADD, bool OUT_BF16>
__global__ __launch_bounds__(256) void gemm_wmma_kernel(
    const void* __restrict__ Av, const float* __restrict__ A2,
    const bf16* __restrict__ BT, const float* __restrict__ bias,
    const float* __restrict__ res, void* __restrict__ Out,
    int M, int N, int K) {
  constexpr int BM = 128, BN = 64, BK = 32, LDB = BK + 8;
  __shared__ bf16 As[2][BM][LDB];
  __shared__ bf16 Bs[2][BN][LDB];

  const int t = threadIdx.x;
  const int lane = t & 31, wave = t >> 5;
  const int wm = wave >> 1, wn = wave & 1;
  const int l16 = lane & 15, lh = lane >> 4;
  const int bm0 = blockIdx.x * BM, bn0 = blockIdx.y * BN;
  const int nIter = K / BK;

  const int bn_ = t >> 2, bc8 = (t & 3) * 8;   // B staging: 1 x 16B per lane
  const int ar = t >> 1, ac16 = (t & 1) * 16;  // A staging: 2 x 16B per lane

  v8f acc[2][2] = {};

  auto issueB = [&](int k0, int buf) {
    async_copy_b128(BT + (size_t)(bn0 + bn_) * K + k0 + bc8,
                    &Bs[buf][bn_][bc8]);
  };

  auto compute = [&](int buf) {
    union FragU { uint4 u[2]; v16bf v; };
    FragU a[2], b[2];
#pragma unroll
    for (int tm = 0; tm < 2; ++tm) {
      // A 16x32 frag: lane row = l16; elems 0..7 -> K=lh*8.., 8..15 -> K=16+lh*8..
      const bf16* p = &As[buf][wm * 32 + tm * 16 + l16][lh * 8];
      a[tm].u[0] = *(const uint4*)p;
      a[tm].u[1] = *(const uint4*)(p + 16);
    }
#pragma unroll
    for (int tn = 0; tn < 2; ++tn) {
      // B 32x16 frag: lane col = l16; elems 0..7 -> K=lh*8.., 8..15 -> K=16+lh*8..
      const bf16* p = &Bs[buf][wn * 32 + tn * 16 + l16][lh * 8];
      b[tn].u[0] = *(const uint4*)p;
      b[tn].u[1] = *(const uint4*)(p + 16);
    }
#pragma unroll
    for (int tm = 0; tm < 2; ++tm)
#pragma unroll
      for (int tn = 0; tn < 2; ++tn)
        acc[tm][tn] = __builtin_amdgcn_wmma_f32_16x16x32_bf16(
            false, a[tm].v, false, b[tn].v, (short)0, acc[tm][tn], false, false);
  };

  if constexpr (!A_F32) {
    // ---- async double-buffered pipeline: copies for tile i+1 fly under
    // ---- the WMMAs of tile i.
    const bf16* A = (const bf16*)Av;
    auto issueA = [&](int k0, int buf) {
      const bf16* g = A + (size_t)(bm0 + ar) * K + k0 + ac16;
      async_copy_b128(g, &As[buf][ar][ac16]);
      async_copy_b128(g + 8, &As[buf][ar][ac16 + 8]);
    };
    issueA(0, 0);
    issueB(0, 0);
    for (int it = 0; it < nIter; ++it) {
      const int buf = it & 1;
      s_wait_asynccnt0();   // my copies for tile `it` have landed
      __syncthreads();      // everyone's have
      if (it + 1 < nIter) { // prefetch next tile into the other buffer
        issueA((it + 1) * BK, buf ^ 1);
        issueB((it + 1) * BK, buf ^ 1);
      }
      compute(buf);
      __syncthreads();
    }
  } else {
    // ---- guarded synchronous A stage (f32 -> bf16 conversion), async B ----
    const float* A = (const float*)Av;
    for (int it = 0; it < nIter; ++it) {
      const int k0 = it * BK;
      issueB(k0, 0);
#pragma unroll
      for (int i = 0; i < 4; ++i) {
        int u = t + i * 256;  // 1024 float4 slots
        int row = u >> 3, c4 = (u & 7) * 4;
        float4 f = make_float4(0.f, 0.f, 0.f, 0.f);
        if (bm0 + row < M) {
          const float* p = A + (size_t)(bm0 + row) * K + k0 + c4;
          f = *(const float4*)p;
          if (ADD2) {
            float4 g = *(const float4*)(A2 + (size_t)(bm0 + row) * K + k0 + c4);
            f.x += g.x; f.y += g.y; f.z += g.z; f.w += g.w;
          }
          if (k0 + BK < K) __builtin_prefetch(p + BK, 0, 1);
        }
        union { bf16 h[4]; uint2 u2; } pk;
        pk.h[0] = f2bf(f.x); pk.h[1] = f2bf(f.y);
        pk.h[2] = f2bf(f.z); pk.h[3] = f2bf(f.w);
        *(uint2*)&As[0][row][c4] = pk.u2;
      }
      s_wait_asynccnt0();
      __syncthreads();
      compute(0);
      __syncthreads();
    }
  }

  // ---- epilogue: D layout = lanes 0-15 rows 0..7, lanes 16-31 rows 8..15 ----
#pragma unroll
  for (int tm = 0; tm < 2; ++tm) {
#pragma unroll
    for (int tn = 0; tn < 2; ++tn) {
      int col = bn0 + wn * 32 + tn * 16 + l16;
      float bv = bias ? bias[col] : 0.0f;
#pragma unroll
      for (int i = 0; i < 8; ++i) {
        int row = bm0 + wm * 32 + tm * 16 + lh * 8 + i;
        if (row < M) {
          float v = acc[tm][tn][i] + bv;
          if (SILU_EP) v = silu(v);
          if (RES_ADD) v += res[(size_t)row * N + col];
          if (OUT_BF16) ((bf16*)Out)[(size_t)row * N + col] = f2bf(v);
          else          ((float*)Out)[(size_t)row * N + col] = v;
        }
      }
    }
  }
}

// ---------------------------------------------------------------------------
// Edge MLP layer 1 (K=3, not worth WMMA): LN(3) -> Linear 3x256 -> SiLU -> bf16
// ---------------------------------------------------------------------------
__global__ __launch_bounds__(256) void edge_mlp1_kernel(
    const float* __restrict__ ea, const float* __restrict__ lg,
    const float* __restrict__ lb, const float* __restrict__ w1,
    const float* __restrict__ b1, bf16* __restrict__ out) {
  int e = blockIdx.x, n = threadIdx.x;
  float a0 = ea[(size_t)e * 3 + 0];
  float a1 = ea[(size_t)e * 3 + 1];
  float a2 = ea[(size_t)e * 3 + 2];
  float mu = (a0 + a1 + a2) * (1.0f / 3.0f);
  float d0 = a0 - mu, d1 = a1 - mu, d2 = a2 - mu;
  float rs = rsqrtf((d0 * d0 + d1 * d1 + d2 * d2) * (1.0f / 3.0f) + 1e-5f);
  float h0 = d0 * rs * lg[0] + lb[0];
  float h1 = d1 * rs * lg[1] + lb[1];
  float h2 = d2 * rs * lg[2] + lb[2];
  float v = h0 * w1[n] + h1 * w1[256 + n] + h2 * w1[512 + n] + b1[n];
  out[(size_t)e * 256 + n] = f2bf(silu(v));
}

// ---------------------------------------------------------------------------
// LayerNorm over 256 cols, f32 in -> bf16 out. One wave32 per row.
// ---------------------------------------------------------------------------
__global__ __launch_bounds__(256) void layernorm256_kernel(
    const float* __restrict__ x, const float* __restrict__ g,
    const float* __restrict__ b, bf16* __restrict__ y, int M) {
  int lane = threadIdx.x & 31;
  int row = blockIdx.x * 8 + (threadIdx.x >> 5);
  if (row >= M) return;
  const float* xp = x + (size_t)row * 256 + lane * 8;
  float4 f0 = *(const float4*)xp;
  float4 f1 = *(const float4*)(xp + 4);
  float vals[8] = {f0.x, f0.y, f0.z, f0.w, f1.x, f1.y, f1.z, f1.w};
  float s = 0.f, ss = 0.f;
#pragma unroll
  for (int j = 0; j < 8; ++j) { s += vals[j]; ss += vals[j] * vals[j]; }
#pragma unroll
  for (int m = 16; m >= 1; m >>= 1) {
    s  += __shfl_xor(s, m, 32);
    ss += __shfl_xor(ss, m, 32);
  }
  float mean = s * (1.0f / 256.0f);
  float var = ss * (1.0f / 256.0f) - mean * mean;
  float rs = rsqrtf(var + 1e-5f);
  union { bf16 h[8]; uint4 u; } pk;
#pragma unroll
  for (int j = 0; j < 8; ++j) {
    int c = lane * 8 + j;
    pk.h[j] = f2bf((vals[j] - mean) * rs * g[c] + b[c]);
  }
  *(uint4*)(y + (size_t)row * 256 + lane * 8) = pk.u;
}

// ---------------------------------------------------------------------------
// Attention: alpha[e,h] = <q[dst], k[src]+ee[e]> / 4 ; segment softmax.
// ---------------------------------------------------------------------------
__global__ __launch_bounds__(256) void attn_alpha_kernel(
    const float* __restrict__ Q, const float* __restrict__ Km,
    const bf16* __restrict__ EE, const int* __restrict__ src_idx,
    const int* __restrict__ dst_idx, float* __restrict__ alpha,
    unsigned* __restrict__ amax, int nEH) {
  int tid = blockIdx.x * 256 + threadIdx.x;
  if (tid >= nEH) return;
  int e = tid >> 4, h = tid & 15;
  int ds = dst_idx[e], sr = src_idx[e];
  const float* qp = Q  + (size_t)ds * 256 + h * 16;
  const float* kp = Km + (size_t)sr * 256 + h * 16;
  const bf16*  ep = EE + (size_t)e  * 256 + h * 16;
  float dot = 0.f;
#pragma unroll
  for (int j = 0; j < 16; ++j) dot += qp[j] * (kp[j] + bf2f(ep[j]));
  float a = dot * 0.25f;  // 1/sqrt(C), C=16
  alpha[tid] = a;
  atomicMax(&amax[ds * 16 + h], f2key(a));
}

__global__ __launch_bounds__(256) void attn_exp_kernel(
    const int* __restrict__ dst_idx, const unsigned* __restrict__ amax,
    float* __restrict__ alpha, float* __restrict__ asum, int nEH) {
  int tid = blockIdx.x * 256 + threadIdx.x;
  if (tid >= nEH) return;
  int e = tid >> 4, h = tid & 15;
  int ds = dst_idx[e];
  float w = __expf(alpha[tid] - key2f(amax[ds * 16 + h]));
  alpha[tid] = w;
  atomicAdd(&asum[ds * 16 + h], w);
}

__global__ __launch_bounds__(256) void attn_scatter_kernel(
    const float* __restrict__ V, const int* __restrict__ src_idx,
    const int* __restrict__ dst_idx, const float* __restrict__ alpha,
    const float* __restrict__ asum, float* __restrict__ att, int nEH) {
  int tid = blockIdx.x * 256 + threadIdx.x;
  if (tid >= nEH) return;
  int e = tid >> 4, h = tid & 15;
  int ds = dst_idx[e], sr = src_idx[e];
  float w = alpha[tid] / (asum[ds * 16 + h] + 1e-16f);
  const float* vp = V + (size_t)sr * 256 + h * 16;
  float* op = att + (size_t)ds * 256 + h * 16;
#pragma unroll
  for (int j = 0; j < 16; ++j) atomicAdd(&op[j], vp[j] * w);
}

// ---------------------------------------------------------------------------
// Host-side launch
// ---------------------------------------------------------------------------
extern "C" void kernel_launch(void* const* d_in, const int* in_sizes, int n_in,
                              void* d_out, int out_size, void* d_ws,
                              size_t ws_size, hipStream_t stream) {
  (void)in_sizes; (void)n_in; (void)out_size; (void)ws_size;
  constexpr int N_SRC = 100000, N_DST = 40000, NE = 400000;
  constexpr int HID = 256, MLP_HID = 1024, IN = 128;

  // ---- inputs (setup_inputs dict order, params in insertion order) ----
  const float* x_src     = (const float*)d_in[0];
  const float* x_dst     = (const float*)d_in[1];
  const float* edge_attr = (const float*)d_in[2];
  const int*   src_idx   = (const int*)d_in[3];
  const int*   dst_idx   = (const int*)d_in[4];
  const float* emb_src_w = (const float*)d_in[5];
  const float* emb_src_b = (const float*)d_in[6];
  const float* emb_dst_w = (const float*)d_in[7];
  const float* emb_dst_b = (const float*)d_in[8];
  const float* e_ln_g = (const float*)d_in[9];
  const float* e_ln_b = (const float*)d_in[10];
  const float* e_w1   = (const float*)d_in[11];
  const float* e_b1   = (const float*)d_in[12];
  const float* e_w2   = (const float*)d_in[13];
  const float* e_b2   = (const float*)d_in[14];
  const float* ln1_g = (const float*)d_in[15];
  const float* ln1_b = (const float*)d_in[16];
  const float* ln2_g = (const float*)d_in[17];
  const float* ln2_b = (const float*)d_in[18];
  const float* wq = (const float*)d_in[19]; const float* bq = (const float*)d_in[20];
  const float* wk = (const float*)d_in[21]; const float* bk = (const float*)d_in[22];
  const float* wv = (const float*)d_in[23]; const float* bv = (const float*)d_in[24];
  const float* ws = (const float*)d_in[25]; const float* bs = (const float*)d_in[26];
  const float* we = (const float*)d_in[27]; const float* be = (const float*)d_in[28];
  const float* wp = (const float*)d_in[29]; const float* bp = (const float*)d_in[30];
  const float* d_ln_g = (const float*)d_in[31];
  const float* d_ln_b = (const float*)d_in[32];
  const float* d_w1 = (const float*)d_in[33]; const float* d_b1 = (const float*)d_in[34];
  const float* d_w2 = (const float*)d_in[35]; const float* d_b2 = (const float*)d_in[36];
  const float* s_ln_g = (const float*)d_in[37];
  const float* s_ln_b = (const float*)d_in[38];
  const float* s_w1 = (const float*)d_in[39]; const float* s_b1 = (const float*)d_in[40];
  const float* s_w2 = (const float*)d_in[41]; const float* s_b2 = (const float*)d_in[42];

  // ---- workspace layout ----
  char* base = (char*)d_ws;
  size_t off = 0;
  auto take = [&](size_t nb) {
    char* p = base + off;
    off += (nb + 255) & ~(size_t)255;
    return p;
  };
  // bf16 weights, pre-transposed [N][K] (except W2B, kept [K][N] = (w2^T)^T)
  bf16* WB = (bf16*)take((size_t)1638400 * 2);
  bf16* WT_esrc = WB + 0;        // [256][128]
  bf16* WT_edst = WB + 32768;    // [256][128]
  bf16* WT_q    = WB + 65536;    // [256][256]
  bf16* WT_k    = WB + 131072;
  bf16* WT_v    = WB + 196608;
  bf16* WT_s    = WB + 262144;
  bf16* WT_p    = WB + 327680;
  bf16* WET     = WB + 393216;   // we^T  (A operand of the fusion GEMM)
  bf16* W2B     = WB + 458752;   // w2    (B^T operand of the fusion GEMM)
  bf16* WT_dw1  = WB + 524288;   // [1024][256]
  bf16* WT_dw2  = WB + 786432;   // [256][1024]
  bf16* WT_sw1  = WB + 1048576;
  bf16* WT_sw2  = WB + 1310720;
  bf16* WT_ee   = WB + 1572864;  // (w2@we)^T, written by the fusion GEMM

  float* BEE  = (float*)take(256 * 4);
  bf16*  HE   = (bf16*)take((size_t)NE * HID * 2);  // h_e; reused: hid_src, hid_dst
  bf16*  EEb  = (bf16*)take((size_t)NE * HID * 2);  // ee (bf16)
  float* S    = (float*)take((size_t)N_SRC * HID * 4);
  float* Dd   = (float*)take((size_t)N_DST * HID * 4);
  bf16*  SN   = (bf16*)take((size_t)N_SRC * HID * 2);  // ln1(s); reused: ln_src
  bf16*  DN   = (bf16*)take((size_t)N_DST * HID * 2);  // ln2(d); reused: ln_out
  float* Qb   = (float*)take((size_t)N_DST * HID * 4);
  float* Kb   = (float*)take((size_t)N_SRC * HID * 4);
  float* Vb   = (float*)take((size_t)N_SRC * HID * 4);
  float* XR   = (float*)take((size_t)N_DST * HID * 4);
  float* ALPH = (float*)take((size_t)NE * 16 * 4);
  unsigned* AMAX = (unsigned*)take((size_t)N_DST * 16 * 4);
  float* ASUM = (float*)take((size_t)N_DST * 16 * 4);
  float* ATT  = (float*)take((size_t)N_DST * HID * 4);
  float* OUTP = (float*)take((size_t)N_DST * HID * 4);

  float* out_src = (float*)d_out;                        // src_new [N_SRC,HID]
  float* out_dst = (float*)d_out + (size_t)N_SRC * HID;  // dst_new [N_DST,HID]

  auto cvt = [&](const float* src, bf16* dst, int n) {
    cvt_f32_bf16_kernel<<<(n + 255) / 256, 256, 0, stream>>>(src, dst, n);
  };
  auto cvtT = [&](const float* src, bf16* dst, int Kd, int Nd) {
    cvt_t_f32_bf16_kernel<<<(Kd * Nd + 255) / 256, 256, 0, stream>>>(src, dst, Kd, Nd);
  };
  auto gdim = [](int M, int N) { return dim3((M + 127) / 128, N / 64); };

  // ---- 1) weight conversion (pre-transposed) ----
  cvtT(emb_src_w, WT_esrc, IN, HID);
  cvtT(emb_dst_w, WT_edst, IN, HID);
  cvtT(wq, WT_q, HID, HID);  cvtT(wk, WT_k, HID, HID);
  cvtT(wv, WT_v, HID, HID);  cvtT(ws, WT_s, HID, HID);
  cvtT(wp, WT_p, HID, HID);  cvtT(we, WET, HID, HID);
  cvt (e_w2, W2B, HID * HID);
  cvtT(d_w1, WT_dw1, HID, MLP_HID);  cvtT(d_w2, WT_dw2, MLP_HID, HID);
  cvtT(s_w1, WT_sw1, HID, MLP_HID);  cvtT(s_w2, WT_sw2, MLP_HID, HID);

  // ---- 2) fused edge weights: W_ee^T = we^T @ w2^T ; BEE = b2@we + be ----
  bias_fuse_kernel<<<1, 256, 0, stream>>>(e_b2, we, be, BEE);
  gemm_wmma_kernel<false, false, false, false, true>
      <<<gdim(HID, HID), 256, 0, stream>>>(WET, nullptr, W2B, nullptr, nullptr,
                                           WT_ee, HID, HID, HID);

  // ---- 3) edge MLP layer 1 (LN + Linear(3->256) + SiLU) ----
  edge_mlp1_kernel<<<NE, 256, 0, stream>>>(edge_attr, e_ln_g, e_ln_b, e_w1,
                                           e_b1, HE);
  // ee = h_e @ W_ee + BEE  (bf16 out)
  gemm_wmma_kernel<false, false, false, false, true>
      <<<gdim(NE, HID), 256, 0, stream>>>(HE, nullptr, WT_ee, BEE, nullptr, EEb,
                                          NE, HID, HID);

  // ---- 4) node embeddings ----
  gemm_wmma_kernel<true, false, false, false, false>
      <<<gdim(N_SRC, HID), 256, 0, stream>>>(x_src, nullptr, WT_esrc, emb_src_b,
                                             nullptr, S, N_SRC, HID, IN);
  gemm_wmma_kernel<true, false, false, false, false>
      <<<gdim(N_DST, HID), 256, 0, stream>>>(x_dst, nullptr, WT_edst, emb_dst_b,
                                             nullptr, Dd, N_DST, HID, IN);

  // ---- 5) pre-attention LayerNorms (bf16 out) ----
  layernorm256_kernel<<<(N_SRC + 7) / 8, 256, 0, stream>>>(S, ln1_g, ln1_b, SN, N_SRC);
  layernorm256_kernel<<<(N_DST + 7) / 8, 256, 0, stream>>>(Dd, ln2_g, ln2_b, DN, N_DST);

  // ---- 6) q/k/v/x_r projections ----
  gemm_wmma_kernel<false, false, false, false, false>
      <<<gdim(N_DST, HID), 256, 0, stream>>>(DN, nullptr, WT_q, bq, nullptr, Qb,
                                             N_DST, HID, HID);
  gemm_wmma_kernel<false, false, false, false, false>
      <<<gdim(N_SRC, HID), 256, 0, stream>>>(SN, nullptr, WT_k, bk, nullptr, Kb,
                                             N_SRC, HID, HID);
  gemm_wmma_kernel<false, false, false, false, false>
      <<<gdim(N_SRC, HID), 256, 0, stream>>>(SN, nullptr, WT_v, bv, nullptr, Vb,
                                             N_SRC, HID, HID);
  gemm_wmma_kernel<false, false, false, false, false>
      <<<gdim(N_DST, HID), 256, 0, stream>>>(DN, nullptr, WT_s, bs, nullptr, XR,
                                             N_DST, HID, HID);

  // ---- 7) src branch (done now so SN/HE can be reused) ----
  layernorm256_kernel<<<(N_SRC + 7) / 8, 256, 0, stream>>>(S, s_ln_g, s_ln_b, SN, N_SRC);
  gemm_wmma_kernel<false, false, true, false, true>
      <<<gdim(N_SRC, MLP_HID), 256, 0, stream>>>(SN, nullptr, WT_sw1, s_b1,
                                                 nullptr, HE, N_SRC, MLP_HID, HID);
  gemm_wmma_kernel<false, false, false, true, false>
      <<<gdim(N_SRC, HID), 256, 0, stream>>>(HE, nullptr, WT_sw2, s_b2, S,
                                             out_src, N_SRC, HID, MLP_HID);

  // ---- 8) attention: segment softmax + scatter ----
  hipMemsetAsync(AMAX, 0, (size_t)N_DST * 16 * 4, stream);  // key 0 == -inf
  hipMemsetAsync(ASUM, 0, (size_t)N_DST * 16 * 4, stream);
  hipMemsetAsync(ATT, 0, (size_t)N_DST * HID * 4, stream);
  const int nEH = NE * 16;
  const int ablk = (nEH + 255) / 256;
  attn_alpha_kernel<<<ablk, 256, 0, stream>>>(Qb, Kb, EEb, src_idx, dst_idx,
                                              ALPH, AMAX, nEH);
  attn_exp_kernel<<<ablk, 256, 0, stream>>>(dst_idx, AMAX, ALPH, ASUM, nEH);
  attn_scatter_kernel<<<ablk, 256, 0, stream>>>(Vb, src_idx, dst_idx, ALPH,
                                                ASUM, ATT, nEH);

  // ---- 9) proj: out = (att + x_r) @ wp + bp + d ----
  gemm_wmma_kernel<true, true, false, true, false>
      <<<gdim(N_DST, HID), 256, 0, stream>>>(ATT, XR, WT_p, bp, Dd, OUTP,
                                             N_DST, HID, HID);

  // ---- 10) dst branch MLP ----
  layernorm256_kernel<<<(N_DST + 7) / 8, 256, 0, stream>>>(OUTP, d_ln_g, d_ln_b,
                                                           DN, N_DST);
  gemm_wmma_kernel<false, false, true, false, true>
      <<<gdim(N_DST, MLP_HID), 256, 0, stream>>>(DN, nullptr, WT_dw1, d_b1,
                                                 nullptr, HE, N_DST, MLP_HID, HID);
  gemm_wmma_kernel<false, false, false, true, false>
      <<<gdim(N_DST, HID), 256, 0, stream>>>(HE, nullptr, WT_dw2, d_b2, OUTP,
                                             out_dst, N_DST, HID, MLP_HID);
}
